// MDTA_8040178778949
// MI455X (gfx1250) — matrix operand order repaired and checked by reading, not compiled, not run on hardware
//
#include <hip/hip_runtime.h>
#include <hip/hip_bf16.h>
#include <cstdint>

// ---------------------------------------------------------------------------
// MDTA (Restormer transposed channel attention) for MI455X / gfx1250.
// B=8, C=192, H=W=128, heads=4, dh=48, HW=16384.
// - all GEMM stages: v_wmma_f32_16x16x32_bf16 (wave32, f32 accumulate)
// - A-tiles (weights): ONE tensor_load_to_lds (TDM) per block, LDS row
//   padding via D# pad fields -> conflict-free ds_load_b128 gathers,
//   K-padding via TDM OOB-zero semantics
// - attention q/k tiles: double-buffered global_load_async_to_lds_b128,
//   s_wait_asynccnt 1 overlaps DMA with WMMA
// - B-tiles: software-pipelined b64 global loads + register transpose
// Pipeline is HBM-bound (~1 GB traffic vs ~45 GFLOP -> ~45us at 23.3 TB/s).
// ---------------------------------------------------------------------------

typedef __attribute__((ext_vector_type(16))) __bf16 v16bf;
typedef __attribute__((ext_vector_type(8)))  float  v8f;
typedef unsigned int tdm4u __attribute__((ext_vector_type(4)));
typedef int          tdm8i __attribute__((ext_vector_type(8)));
typedef int          tdm4i __attribute__((ext_vector_type(4)));

#define HW_   16384
#define CCH   192
#define HEADS 4
#define DH    48
#define BSTRIDE 40   // 80B = 20 dw rows: gcd(20,64)=4 -> conflict-free b128 gathers

static __device__ __forceinline__ unsigned short f2bf(float f) {
    unsigned int u = __float_as_uint(f);
    u += 0x7FFFu + ((u >> 16) & 1u);
    return (unsigned short)(u >> 16);
}
static __device__ __forceinline__ float bf2f(unsigned short h) {
    return __uint_as_float(((unsigned int)h) << 16);
}
static __device__ __forceinline__ float bflo(unsigned int u) { return __uint_as_float(u << 16); }
static __device__ __forceinline__ float bfhi(unsigned int u) { return __uint_as_float(u & 0xFFFF0000u); }

struct alignas(8)  US4 { unsigned short s[4]; };
struct alignas(16) US8 { unsigned short s[8]; };

static __device__ __forceinline__ unsigned lds_off32(const void* p) {
    return (unsigned)(uintptr_t)(const __attribute__((address_space(3))) void*)p;
}
static __device__ __forceinline__ void async_b128(unsigned lds_dst, const void* gsrc) {
    asm volatile("global_load_async_to_lds_b128 %0, %1, off"
                 :: "v"(lds_dst), "v"(gsrc) : "memory");
}
static __device__ __forceinline__ void wait_async0() {
    asm volatile("s_wait_asynccnt 0x0" ::: "memory");
}
static __device__ __forceinline__ void wait_async1() {
    asm volatile("s_wait_asynccnt 0x1" ::: "memory");
}

// One TDM 2-D tile load: ROWSxTILE0 elems (2B), tensor_dim0=KREAL so columns
// >= KREAL read as zero; LDS rows padded +16B per 64 elems (pad_interval=32dw,
// pad_amount=4dw).
static __device__ __forceinline__ void tdm_load_tile(unsigned lds, const void* gsrc,
                                                     unsigned kreal, unsigned tile0,
                                                     unsigned rows) {
    unsigned long long ga = (unsigned long long)(uintptr_t)gsrc;
    tdm4u g0 = { 1u,                                   // count=1 (load, user D#)
                 lds,                                  // lds_addr
                 (unsigned)(ga & 0xFFFFFFFFu),
                 (unsigned)((ga >> 32) & 0x01FFFFFFu) | (2u << 30) };  // type=2
    tdm8i g1 = { (int)((1u << 16) | (1u << 20) | (4u << 22) | (3u << 25)),
                 //    data_size=2B   pad_en      pad_int=32dw  pad_amt=4dw
                 (int)((kreal & 0xFFFFu) << 16),       // tensor_dim0 lo16
                 (int)((kreal >> 16) | ((rows & 0xFFFFu) << 16)),   // d0 hi / d1 lo
                 (int)((tile0 & 0xFFFFu) << 16),       // d1 hi=0, tile_dim0
                 (int)(rows & 0xFFFFu),                // tile_dim1, tile_dim2=0
                 (int)kreal,                           // tensor_dim0_stride lo32
                 0, 0 };
    tdm4i g2 = {0, 0, 0, 0};
    tdm4i g3 = {0, 0, 0, 0};
#if defined(__clang_major__) && __clang_major__ >= 23
    tdm8i g4 = {0, 0, 0, 0, 0, 0, 0, 0};
    __builtin_amdgcn_tensor_load_to_lds(g0, g1, g2, g3, g4, 0);
#else
    __builtin_amdgcn_tensor_load_to_lds(g0, g1, g2, g3, 0);
#endif
}

// A-fragment gather: 16x32 bf16 tile row-major [m][k] in LDS (stride elems).
static __device__ __forceinline__ v16bf gatherA(const unsigned short* lds, int lane, int stride) {
    int m = lane & 15, hi = lane >> 4;
    union { v16bf v; unsigned int u[8]; } r;
    const unsigned int* row = (const unsigned int*)(lds + m * stride);
#pragma unroll
    for (int p = 0; p < 8; ++p) {
        int k = (p < 4 ? 2 * p : 16 + 2 * (p - 4)) + hi * 8;
        r.u[p] = row[k >> 1];
    }
    return r.v;
}
// B-fragment gather: 32x16 tile stored TRANSPOSED [n][k] in LDS.
static __device__ __forceinline__ v16bf gatherB(const unsigned short* lds, int lane, int stride) {
    int n = lane & 15, hi = lane >> 4;
    union { v16bf v; unsigned int u[8]; } r;
    const unsigned int* row = (const unsigned int*)(lds + n * stride);
#pragma unroll
    for (int p = 0; p < 8; ++p) {
        int k = (p < 4 ? 2 * p : 16 + 2 * (p - 4)) + hi * 8;
        r.u[p] = row[k >> 1];
    }
    return r.v;
}

// ---------------------------------------------------------------------------
__global__ void k_f32_to_bf16(const float* __restrict__ in,
                              unsigned short* __restrict__ out, int n) {
    int i = (blockIdx.x * blockDim.x + threadIdx.x) * 4;
    if (i + 3 < n) {
        float4 v = *(const float4*)(in + i);
        US4 o{f2bf(v.x), f2bf(v.y), f2bf(v.z), f2bf(v.w)};
        *(US4*)(out + i) = o;
    } else {
        for (int j = i; j < n; ++j) out[j] = f2bf(in[j]);
    }
}

// ---------------------------------------------------------------------------
// Generic bf16 WMMA GEMM: Out[bz][m][n] = sum_k W[bz][m][k] * X[bz][k][n]
// 256 thr = 8 waves; block tile 16(M) x 128(N); wave = one 16x16 tile.
// A staged once by TDM (wave 0); B software-pipelined per 32-wide K step.
// ---------------------------------------------------------------------------
template<bool OUTF32, int KREAL, int INNER>
__global__ void __launch_bounds__(256)
k_gemm_bf16(const unsigned short* __restrict__ W, long wbs,
            const unsigned short* __restrict__ X, long xsA, long xsB,
            void* __restrict__ Out, long osA, long osB)
{
    constexpr int NN    = HW_;
    constexpr int NK    = (KREAL + 31) / 32;
    constexpr int TILE0 = ((KREAL + 63) / 64) * 64;       // 192 or 64
    constexpr int AST   = TILE0 + (TILE0 / 64) * 8;       // 216 or 72 elems
    constexpr bool PAD  = (KREAL % 32) != 0;

    __shared__ unsigned short lds_a[16 * AST];
    __shared__ unsigned short lds_b[128 * BSTRIDE];

    int tid = threadIdx.x;
    int bz  = blockIdx.z;
    int b   = bz / INNER, ii = bz % INNER;
    const unsigned short* Wb = W + (long)bz * wbs;
    const unsigned short* Xb = X + (long)b * xsA + (long)ii * xsB;
    long obase = (long)b * osA + (long)ii * osB;

    int m0 = blockIdx.y * 16;
    int n0 = blockIdx.x * 128;
    int wave = tid >> 5, lane = tid & 31;

    // ---- whole A tile: one TDM descriptor (wave 0 issues) ----
    if (wave == 0)
        tdm_load_tile(lds_off32(lds_a), Wb + (long)m0 * KREAL, KREAL, TILE0, 16);
    __builtin_amdgcn_s_wait_tensorcnt(0);
    __syncthreads();

    // ---- software-pipelined K loop over B ----
    int n4 = (tid & 31) << 2;            // 4 columns / thread
    int kq = (tid >> 5) << 2;            // 4 k-rows / thread
    const unsigned short* Xcol = Xb + n0 + n4;

    US4 cur[4];
#pragma unroll
    for (int r = 0; r < 4; ++r) {
        if (!PAD || (kq + r) < KREAL) cur[r] = *(const US4*)(Xcol + (long)(kq + r) * NN);
        else cur[r] = US4{0, 0, 0, 0};
    }

    v8f acc = {};
#pragma unroll
    for (int kt = 0; kt < NK; ++kt) {
        if (kt) __syncthreads();
#pragma unroll
        for (int j = 0; j < 4; ++j) {
            US4 col{cur[0].s[j], cur[1].s[j], cur[2].s[j], cur[3].s[j]};
            *(US4*)&lds_b[(n4 + j) * BSTRIDE + kq] = col;
        }
        US4 nxt[4];
        if (kt + 1 < NK) {
            int kb = (kt + 1) << 5;
#pragma unroll
            for (int r = 0; r < 4; ++r) {
                int gk = kb + kq + r;
                if (!PAD || gk < KREAL) nxt[r] = *(const US4*)(Xcol + (long)gk * NN);
                else nxt[r] = US4{0, 0, 0, 0};
            }
        }
        __syncthreads();
        int kb5 = kt << 5;
        v16bf a  = gatherA(lds_a + kb5 + ((kb5 >> 6) << 3), lane, AST);
        v16bf bf = gatherB(lds_b + (wave * 16) * BSTRIDE, lane, BSTRIDE);
        acc = __builtin_amdgcn_wmma_f32_16x16x32_bf16(
            false, a, false, bf, (short)0, acc, false, false);
        if (kt + 1 < NK) {
#pragma unroll
            for (int r = 0; r < 4; ++r) cur[r] = nxt[r];
        }
    }

    int hi = lane >> 4;
    int gn = n0 + wave * 16 + (lane & 15);
#pragma unroll
    for (int r = 0; r < 8; ++r) {
        int gm = m0 + r + hi * 8;
        long off = obase + (long)gm * NN + gn;
        if (OUTF32) ((float*)Out)[off] = acc[r];
        else        ((unsigned short*)Out)[off] = f2bf(acc[r]);
    }
}

// ---------------------------------------------------------------------------
// 3x3 depthwise conv, padding=1, [B*576][128][128] bf16; 8 px / thread.
// ---------------------------------------------------------------------------
__global__ void k_dwconv3x3(const unsigned short* __restrict__ in,
                            const float* __restrict__ w,
                            unsigned short* __restrict__ out)
{
    long seg = (long)blockIdx.x * 256 + threadIdx.x;
    int  xs    = (int)(seg & 15) << 3;
    int  yy    = (int)((seg >> 4) & 127);
    long chimg = seg >> 11;
    int  ch    = (int)(chimg % (3 * CCH));
    const unsigned short* img = in + chimg * HW_;
    const float* wk = w + ch * 9;

    float acc[8];
#pragma unroll
    for (int j = 0; j < 8; ++j) acc[j] = 0.f;

#pragma unroll
    for (int dy = -1; dy <= 1; ++dy) {
        int y = yy + dy;
        if ((unsigned)y >= 128u) continue;
        const unsigned short* row = img + (y << 7);
        uint4 m = *(const uint4*)(row + xs);
        float c[10];
        c[1] = bflo(m.x); c[2] = bfhi(m.x);
        c[3] = bflo(m.y); c[4] = bfhi(m.y);
        c[5] = bflo(m.z); c[6] = bfhi(m.z);
        c[7] = bflo(m.w); c[8] = bfhi(m.w);
        c[0] = (xs > 0)   ? bf2f(row[xs - 1]) : 0.f;
        c[9] = (xs < 120) ? bf2f(row[xs + 8]) : 0.f;
        float w0 = wk[(dy + 1) * 3 + 0];
        float w1 = wk[(dy + 1) * 3 + 1];
        float w2 = wk[(dy + 1) * 3 + 2];
#pragma unroll
        for (int j = 0; j < 8; ++j)
            acc[j] += c[j] * w0 + c[j + 1] * w1 + c[j + 2] * w2;
    }
    US8 o;
#pragma unroll
    for (int j = 0; j < 8; ++j) o.s[j] = f2bf(acc[j]);
    *(US8*)(out + chimg * HW_ + (yy << 7) + xs) = o;
}

// ---------------------------------------------------------------------------
__global__ void k_rownorm(const unsigned short* __restrict__ qkvd,
                          float* __restrict__ inv)
{
    __shared__ float red[256];
    int row = blockIdx.x;
    int b = row / 384, ch = row % 384;
    const unsigned short* p = qkvd + ((long)b * (3 * CCH) + ch) * HW_;
    float s = 0.f;
    for (int i = threadIdx.x * 8; i < HW_; i += 256 * 8) {
        uint4 v = *(const uint4*)(p + i);
        float a0 = bflo(v.x), a1 = bfhi(v.x), a2 = bflo(v.y), a3 = bfhi(v.y);
        float a4 = bflo(v.z), a5 = bfhi(v.z), a6 = bflo(v.w), a7 = bfhi(v.w);
        s += a0*a0 + a1*a1 + a2*a2 + a3*a3 + a4*a4 + a5*a5 + a6*a6 + a7*a7;
    }
    red[threadIdx.x] = s;
    __syncthreads();
    for (int st = 128; st > 0; st >>= 1) {
        if (threadIdx.x < st) red[threadIdx.x] += red[threadIdx.x + st];
        __syncthreads();
    }
    if (threadIdx.x == 0) inv[row] = 1.f / fmaxf(sqrtf(red[0]), 1e-12f);
}

__global__ void k_zero_f32(float* __restrict__ p, int n) {
    int i = blockIdx.x * blockDim.x + threadIdx.x;
    if (i < n) p[i] = 0.f;
}

// ---------------------------------------------------------------------------
// Split-K attention logits with double-buffered async-to-LDS staging.
// S[bz][c][d] += sum_n q[c][n]*k[d][n] over a 512-wide n chunk.
// 512 thr = 16 waves; 4x4 WMMA tiles on padded 64x64 logits.
// ---------------------------------------------------------------------------
__global__ void __launch_bounds__(512)
k_attn_partial(const unsigned short* __restrict__ qkvd,
               float* __restrict__ S)
{
    __shared__ unsigned short lqk[2][2][64 * BSTRIDE];   // [buf][q/k][tile]

    int tid = threadIdx.x;
    int bz  = blockIdx.y;
    int b = bz >> 2, h = bz & 3;
    const unsigned short* qb = qkvd + ((long)b * (3 * CCH) + h * DH) * HW_;
    const unsigned short* kb = qkvd + ((long)b * (3 * CCH) + CCH + h * DH) * HW_;

    // zero pad rows 48..63 of all four tiles once
    for (int i = tid; i < 16 * BSTRIDE; i += 512) {
        lqk[0][0][DH * BSTRIDE + i] = 0; lqk[0][1][DH * BSTRIDE + i] = 0;
        lqk[1][0][DH * BSTRIDE + i] = 0; lqk[1][1][DH * BSTRIDE + i] = 0;
    }

    int wave = tid >> 5, lane = tid & 31;
    int mt = wave >> 2, nt = wave & 3;

    // staging assignment (wave-uniform active mask): 192 16B chunks per tile
    int qk   = (tid < 256) ? 0 : 1;
    int sc   = tid & 255;
    int srow = sc >> 2;
    int scol = (sc & 3) << 3;
    bool act = srow < DH;
    const unsigned short* src =
        (qk ? kb : qb) + (long)srow * HW_ + blockIdx.x * 512 + scol;
    unsigned d0 = lds_off32(&lqk[0][qk][srow * BSTRIDE + scol]);
    unsigned d1 = lds_off32(&lqk[1][qk][srow * BSTRIDE + scol]);

    if (act) async_b128(d0, src);                  // prologue: kt=0 -> buf0

    v8f acc = {};
    for (int kt = 0; kt < 16; ++kt) {
        if (kt < 15) {
            if (act) async_b128((kt & 1) ? d0 : d1, src + (kt + 1) * 32);
            wait_async1();                         // current buffer ready
        } else {
            wait_async0();
        }
        __syncthreads();
        const unsigned short* bq = &lqk[kt & 1][0][0];
        const unsigned short* bk = &lqk[kt & 1][1][0];
        v16bf a  = gatherA(bq + mt * 16 * BSTRIDE, lane, BSTRIDE);
        v16bf bf = gatherB(bk + nt * 16 * BSTRIDE, lane, BSTRIDE);
        acc = __builtin_amdgcn_wmma_f32_16x16x32_bf16(
            false, a, false, bf, (short)0, acc, false, false);
        __syncthreads();                           // readers done before overwrite
    }

    int hi = lane >> 4;
    int d = nt * 16 + (lane & 15);
    if (d < DH) {
#pragma unroll
        for (int r = 0; r < 8; ++r) {
            int c = mt * 16 + r + hi * 8;
            if (c < DH) atomicAdd(&S[(long)bz * (DH * DH) + c * DH + d], acc[r]);
        }
    }
}

// ---------------------------------------------------------------------------
__global__ void k_softmax(const float* __restrict__ S,
                          const float* __restrict__ inv,
                          const float* __restrict__ temp,
                          unsigned short* __restrict__ attn)
{
    int bz = blockIdx.x;
    int b = bz >> 2, h = bz & 3;
    int r = threadIdx.x;
    if (r >= DH) return;
    const float* inb = inv + b * (2 * CCH);
    float iq = inb[h * DH + r];
    float t  = temp[h];
    const float* row = S + (long)bz * (DH * DH) + r * DH;
    float s[DH];
    float mx = -3.4e38f;
#pragma unroll 8
    for (int j = 0; j < DH; ++j) {
        float v = row[j] * iq * inb[CCH + h * DH + j] * t;
        s[j] = v;
        mx = fmaxf(mx, v);
    }
    float sum = 0.f;
#pragma unroll 8
    for (int j = 0; j < DH; ++j) { s[j] = __expf(s[j] - mx); sum += s[j]; }
    float rs = 1.f / sum;
    unsigned short* arow = attn + (long)bz * (DH * DH) + r * DH;
#pragma unroll 8
    for (int j = 0; j < DH; ++j) arow[j] = f2bf(s[j] * rs);
}

// ---------------------------------------------------------------------------
extern "C" void kernel_launch(void* const* d_in, const int* in_sizes, int n_in,
                              void* d_out, int out_size, void* d_ws, size_t ws_size,
                              hipStream_t stream) {
    (void)in_sizes; (void)n_in; (void)out_size; (void)ws_size;
    const float* x      = (const float*)d_in[0];
    const float* qkv_w  = (const float*)d_in[1];
    const float* dw_w   = (const float*)d_in[2];
    const float* proj_w = (const float*)d_in[3];
    const float* temp   = (const float*)d_in[4];

    const long B = 8;
    const long xN   = B * CCH * HW_;
    const long qkvN = B * 3 * CCH * HW_;

    char* ws = (char*)d_ws;
    size_t cur = 0;
    auto alloc = [&](size_t bytes) -> void* {
        void* p = ws + cur;
        cur = (cur + bytes + 255) & ~(size_t)255;
        return p;
    };
    unsigned short* xb   = (unsigned short*)alloc(xN * 2);
    unsigned short* wq   = (unsigned short*)alloc(576 * 192 * 2);
    unsigned short* wp   = (unsigned short*)alloc(192 * 192 * 2);
    unsigned short* qkv  = (unsigned short*)alloc(qkvN * 2);
    unsigned short* qkvd = (unsigned short*)alloc(qkvN * 2);
    float*          invn = (float*)alloc(B * 2 * CCH * 4);
    float*          S    = (float*)alloc(32 * DH * DH * 4);
    unsigned short* attn = (unsigned short*)alloc(32 * DH * DH * 2);
    unsigned short* ao   = xb;   // alias: x dead after qkv GEMM

    k_f32_to_bf16<<<dim3((unsigned)(xN / 4 / 256)), 256, 0, stream>>>(x, xb, (int)xN);
    k_f32_to_bf16<<<dim3(576 * 192 / 4 / 256), 256, 0, stream>>>(qkv_w, wq, 576 * 192);
    k_f32_to_bf16<<<dim3(36), 256, 0, stream>>>(proj_w, wp, 192 * 192);

    // qkv 1x1 conv: M=576, K=192
    k_gemm_bf16<false, 192, 1><<<dim3(HW_ / 128, 576 / 16, (unsigned)B), 256, 0, stream>>>(
        wq, 0L, xb, (long)CCH * HW_, 0L,
        qkv, (long)3 * CCH * HW_, 0L);

    k_dwconv3x3<<<dim3((unsigned)(qkvN / 8 / 256)), 256, 0, stream>>>(qkv, dw_w, qkvd);

    k_rownorm<<<dim3((unsigned)(B * 2 * CCH)), 256, 0, stream>>>(qkvd, invn);

    k_zero_f32<<<dim3((32 * DH * DH + 255) / 256), 256, 0, stream>>>(S, 32 * DH * DH);
    k_attn_partial<<<dim3(HW_ / 512, 32), 512, 0, stream>>>(qkvd, S);

    k_softmax<<<dim3(32), 64, 0, stream>>>(S, invn, temp, attn);

    // attn @ v: per-(b,h) 48x48 weights, K=48 (TDM zero-pads to 64)
    k_gemm_bf16<false, 48, 4><<<dim3(HW_ / 128, DH / 16, 32), 256, 0, stream>>>(
        attn, (long)(DH * DH),
        qkvd + (long)2 * CCH * HW_, (long)3 * CCH * HW_, (long)DH * HW_,
        ao, (long)CCH * HW_, (long)DH * HW_);

    // proj 1x1 conv -> f32 output
    k_gemm_bf16<true, 192, 1><<<dim3(HW_ / 128, CCH / 16, (unsigned)B), 256, 0, stream>>>(
        wp, 0L, ao, (long)CCH * HW_, 0L,
        d_out, (long)CCH * HW_, 0L);
}
